// SlidingWindowAttention_40475771797971
// MI455X (gfx1250) — matrix operand order repaired and checked by reading, not compiled, not run on hardware
//
#include <hip/hip_runtime.h>

// MI455X (gfx1250) sliding-window attention, fused QKV-proj + banded softmax + PV,
// followed by output projection GEMM. All matmuls via v_wmma_f32_16x16x32_f16.

typedef __attribute__((ext_vector_type(16))) _Float16 v16h;
typedef __attribute__((ext_vector_type(8)))  float    v8f;

#define SEQ   4096
#define EMB   512
#define HEADS 8
#define HD    64
#define WIN   64
#define NC    (SEQ / WIN)     // 64 chunks
#define WROWS 192             // 3*WIN key/value window rows

__device__ __forceinline__ v8f zero8() {
    v8f z = {0.f, 0.f, 0.f, 0.f, 0.f, 0.f, 0.f, 0.f};
    return z;
}

// A-operand K order (matches b_kmap): lanes 0-15 hold K 0..15 (V v -> K 2v,2v+1),
// lanes 16-31 hold K 16..31.
__device__ __forceinline__ int a_kmap(int v, int half) {
    return 2 * v + 16 * half;
}
// B-operand: lanes 0-15 hold K 0..15 (V v -> K 2v,2v+1), lanes 16-31 hold K 16..31.
__device__ __forceinline__ int b_kmap(int v, int half) {
    return 2 * v + 16 * half;
}

__device__ __forceinline__ v8f wmma16(v16h a, v16h b, v8f c) {
    return __builtin_amdgcn_wmma_f32_16x16x32_f16(
        /*neg_a=*/false, a, /*neg_b=*/false, b,
        /*c_mod=*/(short)0, c, /*reuse_a=*/false, /*reuse_b=*/false);
}

// A tile (16xK slice at row tile m0, k offset k0) from row-major f16 LDS.
__device__ __forceinline__ v16h lds_loadA(const _Float16* S, int ld, int m0, int k0) {
    const int lane = threadIdx.x & 31;
    const int row  = m0 + (lane & 15);
    const int hlf  = lane >> 4;
    v16h a;
    #pragma unroll
    for (int v = 0; v < 8; ++v) {
        int k = k0 + a_kmap(v, hlf);
        a[2 * v]     = S[row * ld + k];
        a[2 * v + 1] = S[row * ld + k + 1];
    }
    return a;
}

// B tile where the LDS array Mat[n][k] provides B[k][n] = Mat[n][k].
__device__ __forceinline__ v16h lds_loadB(const _Float16* S, int ld, int n0, int k0) {
    const int lane = threadIdx.x & 31;
    const int row  = n0 + (lane & 15);
    const int hlf  = lane >> 4;
    v16h b;
    #pragma unroll
    for (int v = 0; v < 8; ++v) {
        int k = k0 + b_kmap(v, hlf);
        b[2 * v]     = S[row * ld + k];
        b[2 * v + 1] = S[row * ld + k + 1];
    }
    return b;
}

// A tile gathered directly from global f32 (converted to f16). `row` is the
// caller-resolved absolute row for this lane.
__device__ __forceinline__ v16h g_loadA_f32(const float* __restrict__ X, int ld,
                                            int row, int k0) {
    const int lane = threadIdx.x & 31;
    const int hlf  = lane >> 4;
    const float* p = X + (size_t)row * ld + k0;
    v16h a;
    #pragma unroll
    for (int v = 0; v < 8; ++v) {
        int k = a_kmap(v, hlf);
        float2 f = *(const float2*)(p + k);   // k even -> 8B aligned
        a[2 * v]     = (_Float16)f.x;
        a[2 * v + 1] = (_Float16)f.y;
    }
    return a;
}

// B tile for y = x @ W^T : B[k][n] = W[n][k], W row-major (f32, converted).
__device__ __forceinline__ v16h g_loadB_wt(const float* __restrict__ W, int ld,
                                           int n0, int k0) {
    const int lane = threadIdx.x & 31;
    const int row  = n0 + (lane & 15);
    const int hlf  = lane >> 4;
    const float* p = W + (size_t)row * ld + k0;
    v16h b;
    #pragma unroll
    for (int v = 0; v < 8; ++v) {
        int k = b_kmap(v, hlf);
        float2 f = *(const float2*)(p + k);
        b[2 * v]     = (_Float16)f.x;
        b[2 * v + 1] = (_Float16)f.y;
    }
    return b;
}

// ---------------------------------------------------------------------------
// Kernel A: one block per (batch n, head h, chunk c). 4 waves (128 threads).
// Fused: per-head Q/K/V projection -> banded scores -> softmax -> P@V.
// ---------------------------------------------------------------------------
__global__ __launch_bounds__(128) void swa_attn_kernel(
    const float* __restrict__ value, const float* __restrict__ key_in,
    const float* __restrict__ query,
    const float* __restrict__ w_v, const float* __restrict__ w_k,
    const float* __restrict__ w_q,
    float* __restrict__ out_pre)
{
    __shared__ _Float16 sQ [WIN   * HD];     //  8 KB  projected Q,  [m][d]
    __shared__ _Float16 sK [WROWS * HD];     // 24 KB  projected K,  [win_row][d]
    __shared__ _Float16 sVt[HD * WROWS];     // 24 KB  projected V^T, [d][win_row]
    __shared__ _Float16 sP [WIN * WROWS];    // 24 KB  attention probs, [m][j]

    const int lane = threadIdx.x & 31;
    const int wv   = threadIdx.x >> 5;       // wave id 0..3
    const int ln   = lane & 15;
    const int hlf  = lane >> 4;

    const int c  = blockIdx.x & (NC - 1);
    const int hh = (blockIdx.x >> 6) & (HEADS - 1);
    const int nb = blockIdx.x >> 9;

    const size_t headOff = (size_t)nb * SEQ * EMB + (size_t)hh * HD;
    const float* Xq = query  + headOff;
    const float* Xk = key_in + headOff;
    const float* Xv = value  + headOff;

    const int m0 = 16 * wv;                  // this wave's 16 query rows in chunk

    // ---- Q projection: q = x @ w_q^T for rows [c*64+m0, +16) ----
    {
        const int grow = c * WIN + m0 + ln;  // always in range
        v16h a0 = g_loadA_f32(Xq, EMB, grow, 0);
        v16h a1 = g_loadA_f32(Xq, EMB, grow, 32);
        #pragma unroll
        for (int tn = 0; tn < 4; ++tn) {
            v8f acc = zero8();
            acc = wmma16(a0, g_loadB_wt(w_q, HD, 16 * tn, 0),  acc);
            acc = wmma16(a1, g_loadB_wt(w_q, HD, 16 * tn, 32), acc);
            #pragma unroll
            for (int r = 0; r < 8; ++r)      // C layout: m=r+8*hlf, n=ln
                sQ[(m0 + r + 8 * hlf) * HD + 16 * tn + ln] = (_Float16)acc[r];
        }
    }

    // ---- K/V projection over the 192-row window; zero OOB rows (tile-uniform) ----
    #pragma unroll
    for (int ti = 0; ti < 3; ++ti) {
        const int wr0 = 48 * wv + 16 * ti;          // window row tile base
        const int g0  = c * WIN - WIN + wr0;        // global row of tile base
        const bool valid = (g0 >= 0) && (g0 < SEQ); // uniform for whole 16-row tile
        if (valid) {
            const int grow = g0 + ln;
            v16h ak0 = g_loadA_f32(Xk, EMB, grow, 0);
            v16h ak1 = g_loadA_f32(Xk, EMB, grow, 32);
            v16h av0 = g_loadA_f32(Xv, EMB, grow, 0);
            v16h av1 = g_loadA_f32(Xv, EMB, grow, 32);
            #pragma unroll
            for (int tn = 0; tn < 4; ++tn) {
                v8f acck = zero8(), accv = zero8();
                acck = wmma16(ak0, g_loadB_wt(w_k, HD, 16 * tn, 0),  acck);
                acck = wmma16(ak1, g_loadB_wt(w_k, HD, 16 * tn, 32), acck);
                accv = wmma16(av0, g_loadB_wt(w_v, HD, 16 * tn, 0),  accv);
                accv = wmma16(av1, g_loadB_wt(w_v, HD, 16 * tn, 32), accv);
                #pragma unroll
                for (int r = 0; r < 8; ++r) {
                    const int m = wr0 + r + 8 * hlf;   // window row
                    const int n = 16 * tn + ln;        // d
                    sK [m * HD + n]     = (_Float16)acck[r];
                    sVt[n * WROWS + m]  = (_Float16)accv[r];
                }
            }
        } else {
            #pragma unroll
            for (int tn = 0; tn < 4; ++tn)
                #pragma unroll
                for (int r = 0; r < 8; ++r) {
                    const int m = wr0 + r + 8 * hlf;
                    const int n = 16 * tn + ln;
                    sK [m * HD + n]    = (_Float16)0.f;
                    sVt[n * WROWS + m] = (_Float16)0.f;
                }
        }
    }
    __syncthreads();

    // ---- Scores S = Qp @ Kp^T : 16x192 per wave, K-dim = 64 ----
    v8f acc[12];
    #pragma unroll
    for (int t = 0; t < 12; ++t) acc[t] = zero8();
    #pragma unroll
    for (int ks = 0; ks < 2; ++ks) {
        v16h a = lds_loadA(sQ, HD, m0, 32 * ks);
        #pragma unroll
        for (int t = 0; t < 12; ++t)
            acc[t] = wmma16(a, lds_loadB(sK, HD, 16 * t, 32 * ks), acc[t]);
    }

    // ---- Banded softmax (band: j in [mq, mq+128]); padded keys give logit 0 ----
    #pragma unroll
    for (int r = 0; r < 8; ++r) {
        const int mq = m0 + r + 8 * hlf;             // query row in chunk (0..63)
        float vals[12];
        float mx = -3.0e38f;
        #pragma unroll
        for (int t = 0; t < 12; ++t) {
            const int j = 16 * t + ln;               // window column for this lane
            const bool inband = (j >= mq) && (j <= mq + 128);
            float x = inband ? acc[t][r] : -3.0e38f;
            vals[t] = x;
            mx = fmaxf(mx, x);
        }
        #pragma unroll
        for (int s = 1; s < 16; s <<= 1)             // row lives in one half-wave
            mx = fmaxf(mx, __shfl_xor(mx, s, 32));
        float sum = 0.f;
        #pragma unroll
        for (int t = 0; t < 12; ++t) {
            float p = (vals[t] > -1.0e38f) ? __expf(vals[t] - mx) : 0.f;
            vals[t] = p;
            sum += p;
        }
        #pragma unroll
        for (int s = 1; s < 16; s <<= 1)
            sum += __shfl_xor(sum, s, 32);
        const float inv = 1.0f / sum;
        #pragma unroll
        for (int t = 0; t < 12; ++t)
            sP[mq * WROWS + 16 * t + ln] = (_Float16)(vals[t] * inv);
    }
    __syncthreads();

    // ---- O = P @ Vp : 16x64 per wave, K-dim = 192 ----
    v8f oacc[4];
    #pragma unroll
    for (int t = 0; t < 4; ++t) oacc[t] = zero8();
    #pragma unroll
    for (int ks = 0; ks < 6; ++ks) {
        v16h a = lds_loadA(sP, WROWS, m0, 32 * ks);
        #pragma unroll
        for (int td = 0; td < 4; ++td)
            oacc[td] = wmma16(a, lds_loadB(sVt, WROWS, 16 * td, 32 * ks), oacc[td]);
    }

    float* O = out_pre + (size_t)nb * SEQ * EMB + (size_t)(c * WIN) * EMB + hh * HD;
    #pragma unroll
    for (int td = 0; td < 4; ++td)
        #pragma unroll
        for (int r = 0; r < 8; ++r) {
            const int m = m0 + r + 8 * hlf;
            const int n = 16 * td + ln;
            O[(size_t)m * EMB + n] = oacc[td][r];
        }
}

// ---------------------------------------------------------------------------
// Kernel B: out = out_pre @ w_o^T + b_o.  (16384 x 512) @ (512 x 512)^T.
// One block = 64x64 output tile, 4 waves, each wave 16x64, K = 512.
// ---------------------------------------------------------------------------
__global__ __launch_bounds__(128) void swa_outproj_kernel(
    const float* __restrict__ X,      // out_pre, (N*L) x 512
    const float* __restrict__ w_o,    // 512 x 512
    const float* __restrict__ b_o,    // 512
    float* __restrict__ out)
{
    const int lane = threadIdx.x & 31;
    const int wv   = threadIdx.x >> 5;
    const int ln   = lane & 15;
    const int hlf  = lane >> 4;

    const int colBlk = blockIdx.x & 7;        // 8 column blocks of 64
    const int rowBlk = blockIdx.x >> 3;       // 256 row blocks of 64
    const int row0 = rowBlk * 64 + wv * 16;
    const int n0   = colBlk * 64;

    v8f acc[4];
    #pragma unroll
    for (int t = 0; t < 4; ++t) acc[t] = zero8();

    for (int ks = 0; ks < 16; ++ks) {
        v16h a = g_loadA_f32(X, EMB, row0 + ln, 32 * ks);
        #pragma unroll
        for (int tn = 0; tn < 4; ++tn)
            acc[tn] = wmma16(a, g_loadB_wt(w_o, EMB, n0 + 16 * tn, 32 * ks), acc[tn]);
    }

    #pragma unroll
    for (int tn = 0; tn < 4; ++tn) {
        const int n = n0 + 16 * tn + ln;
        const float bias = b_o[n];
        #pragma unroll
        for (int r = 0; r < 8; ++r) {
            const int m = row0 + r + 8 * hlf;
            out[(size_t)m * EMB + n] = acc[tn][r] + bias;
        }
    }
}

extern "C" void kernel_launch(void* const* d_in, const int* in_sizes, int n_in,
                              void* d_out, int out_size, void* d_ws, size_t ws_size,
                              hipStream_t stream) {
    // setup_inputs() order: value, key_in, query, w_v, w_k, w_q, w_o, b_o (all f32)
    const float* value  = (const float*)d_in[0];
    const float* key_in = (const float*)d_in[1];
    const float* query  = (const float*)d_in[2];
    const float* w_v    = (const float*)d_in[3];
    const float* w_k    = (const float*)d_in[4];
    const float* w_q    = (const float*)d_in[5];
    const float* w_o    = (const float*)d_in[6];
    const float* b_o    = (const float*)d_in[7];
    float* out = (float*)d_out;

    float* out_pre = (float*)d_ws;            // N*L*E f32 = 32 MiB scratch

    dim3 gridA(4 * HEADS * NC);               // 2048 blocks: (n, h, chunk)
    swa_attn_kernel<<<gridA, 128, 0, stream>>>(value, key_in, query,
                                               w_v, w_k, w_q, out_pre);

    dim3 gridB((4 * SEQ / 64) * (EMB / 64));  // 2048 blocks: 64x64 tiles
    swa_outproj_kernel<<<gridB, 128, 0, stream>>>(out_pre, w_o, b_o, out);
}
